// DistanceLayer_31696858644781
// MI455X (gfx1250) — matrix operand order repaired and checked by reading, not compiled, not run on hardware
//
#include <hip/hip_runtime.h>
#include <hip/hip_bf16.h>

typedef __attribute__((ext_vector_type(16))) _Float16 v16h;
typedef __attribute__((ext_vector_type(8)))  float    v8f;

#define NB 256  // stage-1 blocks (fixed -> deterministic reduction tree)

// ---------------------------------------------------------------------------
// Stage 1: stream features[N,128] once; per-block partial column sums (128)
// and partial sum of squares (1) into workspace.
// ws layout (floats): [0, NB*128)       partial column sums
//                     [NB*128, NB*129)  partial square sums
//                     [NB*129, +129)    final {fbar[128], meanf2}
// ---------------------------------------------------------------------------
__global__ __launch_bounds__(256) void feat_reduce_stage1(
    const float* __restrict__ f, int N, float* __restrict__ ws) {
    __shared__ float4 s4[256];
    __shared__ float  ss[256];

    const int t  = threadIdx.x;
    const int c  = t & 31;   // which float4 of the 128-wide row (32 per row)
    const int rl = t >> 5;   // row-within-block (8 rows per block step)

    const float4* f4 = (const float4*)f;

    float4 acc = make_float4(0.f, 0.f, 0.f, 0.f);
    float  sq  = 0.f;
    for (int row = blockIdx.x * 8 + rl; row < N; row += NB * 8) {
        // speculative prefetch of the next strip (gfx1250 global_prefetch_b8)
        __builtin_prefetch(&f4[(size_t)(row + NB * 8) * 32 + c], 0, 0);
        float4 v = f4[(size_t)row * 32 + c];
        acc.x += v.x; acc.y += v.y; acc.z += v.z; acc.w += v.w;
        sq += v.x * v.x + v.y * v.y + v.z * v.z + v.w * v.w;
    }
    s4[t] = acc;
    ss[t] = sq;
    __syncthreads();

    if (t < 32) {
        float4 a = s4[t];
        for (int j = 1; j < 8; ++j) {
            float4 v = s4[t + 32 * j];
            a.x += v.x; a.y += v.y; a.z += v.z; a.w += v.w;
        }
        ((float4*)ws)[blockIdx.x * 32 + t] = a;  // 128 floats per block
    }
    if (t == 0) {
        float s = 0.f;
        for (int j = 0; j < 256; ++j) s += ss[j];
        ws[NB * 128 + blockIdx.x] = s;
    }
}

// ---------------------------------------------------------------------------
// Stage 2: fold NB partials -> fbar[128] = colsum/N, meanf2 = sqsum/N
// ---------------------------------------------------------------------------
__global__ __launch_bounds__(128) void feat_reduce_stage2(
    const float* __restrict__ ws, float* __restrict__ red, int N) {
    const int d = threadIdx.x;  // 0..127
    float s = 0.f;
    for (int b = 0; b < NB; ++b) s += ws[b * 128 + d];
    red[d] = s / (float)N;
    if (d == 0) {
        float q = 0.f;
        for (int b = 0; b < NB; ++b) q += ws[NB * 128 + b];
        red[128] = q / (float)N;
    }
}

// ---------------------------------------------------------------------------
// Stage 3: out[b] = ||x_b||^2 + meanf2 - 2 * dot(x_b, fbar)
// One wave (32 lanes) per 16-row tile of x; dot products via
// v_wmma_f32_16x16x32_f16 with fbar broadcast across all 16 B-columns.
// A (16x32 f16): lane m = lane&15, half = lane>>4,
//   VGPR v -> K = (v<4 ? 2v : 16+2(v-4)) + 8*half
// B (32x16 f16): VGPR v -> K = (v<4 ? 2v : 16+2(v-4)) + 8*half (all N columns identical)
// D (16x16 f32): lane -> N = lane&15, VGPR v -> M = v + 8*half
// ---------------------------------------------------------------------------
__global__ __launch_bounds__(32) void gemv_wmma(
    const float* __restrict__ x, const float* __restrict__ red,
    float* __restrict__ out) {
    __shared__ float x2s[16];

    const int lane = threadIdx.x;
    const int tile = blockIdx.x;     // 16 rows of x per tile
    const int m    = lane & 15;
    const int half = lane >> 4;

    // Row norms ||x_r||^2 in fp32 (lanes 0..15, one row each).
    if (lane < 16) {
        const float4* xr = (const float4*)(x + (size_t)(tile * 16 + lane) * 128);
        float s = 0.f;
        for (int i = 0; i < 32; ++i) {
            float4 v = xr[i];
            s += v.x * v.x + v.y * v.y + v.z * v.z + v.w * v.w;
        }
        x2s[lane] = s;
    }
    __syncthreads();

    const float* xr = x + (size_t)(tile * 16 + m) * 128;

    v8f c = {};
    for (int chunk = 0; chunk < 4; ++chunk) {   // K = 128 in 4 steps of 32
        const int kbase = chunk * 32;
        v16h a, b;
        for (int v = 0; v < 8; ++v) {
            const int ka = (v < 4 ? 2 * v : 16 + 2 * (v - 4)) + 8 * half;
            a[2 * v]     = (_Float16)xr[kbase + ka];
            a[2 * v + 1] = (_Float16)xr[kbase + ka + 1];
            const int kb = (v < 4 ? 2 * v : 16 + 2 * (v - 4)) + 8 * half;
            b[2 * v]     = (_Float16)red[kbase + kb];
            b[2 * v + 1] = (_Float16)red[kbase + kb + 1];
        }
        c = __builtin_amdgcn_wmma_f32_16x16x32_f16(
            /*neg_a=*/false, a, /*neg_b=*/false, b,
            /*c_mod=*/(short)0, c, /*reuse_a=*/false, /*reuse_b=*/false);
    }

    // All 16 D-columns are identical; take column N==0 (lanes 0 and 16).
    const float meanf2 = red[128];
    if (m == 0) {
        for (int v = 0; v < 8; ++v) {
            const int row = v + 8 * half;
            out[tile * 16 + row] = x2s[row] + meanf2 - 2.0f * c[v];
        }
    }
}

extern "C" void kernel_launch(void* const* d_in, const int* in_sizes, int n_in,
                              void* d_out, int out_size, void* d_ws, size_t ws_size,
                              hipStream_t stream) {
    const float* x        = (const float*)d_in[0];   // [B,128]
    const float* features = (const float*)d_in[1];   // [N,128]
    float*       out      = (float*)d_out;           // [B]
    float*       ws       = (float*)d_ws;

    const int B = in_sizes[0] / 128;
    const int N = in_sizes[1] / 128;

    float* red = ws + NB * 129;  // {fbar[128], meanf2}

    feat_reduce_stage1<<<NB, 256, 0, stream>>>(features, N, ws);
    feat_reduce_stage2<<<1, 128, 0, stream>>>(ws, red, N);
    gemv_wmma<<<B / 16, 32, 0, stream>>>(x, red, out);
}
